// kMeans_69406671503893
// MI455X (gfx1250) — compile-verified, matrix operand order checked by reading, not run on hardware
//
#include <hip/hip_runtime.h>
#include <math.h>

typedef __attribute__((ext_vector_type(16))) __bf16 v16bf;
typedef __attribute__((ext_vector_type(8)))  __bf16 v8bf;
typedef __attribute__((ext_vector_type(8)))  float  v8f;

#define DIM         1024
#define BLOCK_M     32
#define WAVES       8
#define NT_PER_WAVE 8     // 64 n-tiles / 8 waves

// ---------------------------------------------------------------------------
// fp32 -> bf16 hi + bf16 residual helpers (bf16x2 split: ~16-bit mantissa
// accuracy when combined with 3 WMMA passes hi*hi + hi*lo + lo*hi)
// ---------------------------------------------------------------------------
__device__ __forceinline__ void split1(float f, __bf16& h, __bf16& l) {
  h = (__bf16)f;
  l = (__bf16)(f - (float)h);
}

// ---------------------------------------------------------------------------
// Pass 1a: split x (flat) into bf16 hi/lo planes. 8 elements / thread.
// ---------------------------------------------------------------------------
__global__ __launch_bounds__(256)
void split_x_kernel(const float* __restrict__ in,
                    __bf16* __restrict__ hi, __bf16* __restrict__ lo) {
  const size_t t = ((size_t)blockIdx.x * blockDim.x + threadIdx.x) * 8;
  const float4 a = *(const float4*)(in + t);
  const float4 b = *(const float4*)(in + t + 4);
  const float v[8] = {a.x, a.y, a.z, a.w, b.x, b.y, b.z, b.w};
  v8bf h, l;
#pragma unroll
  for (int i = 0; i < 8; ++i) { __bf16 hh, ll; split1(v[i], hh, ll); h[i] = hh; l[i] = ll; }
  *(v8bf*)(hi + t) = h;
  *(v8bf*)(lo + t) = l;
}

// ---------------------------------------------------------------------------
// Pass 1b: split centers into bf16 hi/lo planes AND compute ||c_row||^2 in
// fp32. One wave per row; 8 elements per lane per step.
// ---------------------------------------------------------------------------
__global__ __launch_bounds__(32)
void split_centers_kernel(const float* __restrict__ c,
                          __bf16* __restrict__ hi, __bf16* __restrict__ lo,
                          float* __restrict__ csq) {
  const int row  = blockIdx.x;
  const int lane = threadIdx.x;
  const size_t base = (size_t)row * DIM;
  float s0 = 0.f, s1 = 0.f, s2 = 0.f, s3 = 0.f;
#pragma unroll
  for (int i = 0; i < DIM; i += 32 * 8) {
    const size_t t = base + i + lane * 8;
    const float4 a = *(const float4*)(c + t);
    const float4 b = *(const float4*)(c + t + 4);
    const float v[8] = {a.x, a.y, a.z, a.w, b.x, b.y, b.z, b.w};
    v8bf h, l;
#pragma unroll
    for (int k = 0; k < 8; ++k) { __bf16 hh, ll; split1(v[k], hh, ll); h[k] = hh; l[k] = ll; }
    *(v8bf*)(hi + t) = h;
    *(v8bf*)(lo + t) = l;
    s0 = fmaf(v[0], v[0], fmaf(v[4], v[4], s0));
    s1 = fmaf(v[1], v[1], fmaf(v[5], v[5], s1));
    s2 = fmaf(v[2], v[2], fmaf(v[6], v[6], s2));
    s3 = fmaf(v[3], v[3], fmaf(v[7], v[7], s3));
  }
  float s = (s0 + s1) + (s2 + s3);
#pragma unroll
  for (int off = 1; off < 32; off <<= 1) s += __shfl_xor(s, off, 32);
  if (lane == 0) csq[row] = s;
}

// ---------------------------------------------------------------------------
// Fragment load from a pre-split bf16 plane: lane-half K-pattern
// (K = halfOff..halfOff+7 and 16+halfOff..16+halfOff+7), two b128 loads.
// ---------------------------------------------------------------------------
__device__ __forceinline__ v16bf load_frag_bf16(const __bf16* __restrict__ plane,
                                                int row, int d0, int halfOff) {
  const __bf16* p = plane + (size_t)row * DIM + d0 + halfOff;
  const v8bf c0 = *(const v8bf*)(p);
  const v8bf c1 = *(const v8bf*)(p + 16);
  return __builtin_shufflevector(c0, c1, 0, 1, 2, 3, 4, 5, 6, 7,
                                         8, 9, 10, 11, 12, 13, 14, 15);
}

// ---------------------------------------------------------------------------
// Fast path GEMM + fused log-softmax: pure vmem + wmma main loop.
// ---------------------------------------------------------------------------
__global__ __launch_bounds__(256)
void gemm_lsm_bf16(const __bf16* __restrict__ xh, const __bf16* __restrict__ xl,
                   const __bf16* __restrict__ ch, const __bf16* __restrict__ cl,
                   const float* __restrict__ csq_g,
                   float* __restrict__ out) {
  __shared__ float red[2][WAVES][BLOCK_M];

  const int tid     = threadIdx.x;
  const int wave    = tid >> 5;
  const int lane    = tid & 31;
  const int lrow    = lane & 15;
  const int halfOff = (lane < 16) ? 0 : 8;
  const int half    = (lane < 16) ? 0 : 1;
  const int row0    = blockIdx.x * BLOCK_M;

  v8f acc[2][NT_PER_WAVE];
  const v8f vzero = {0.f, 0.f, 0.f, 0.f, 0.f, 0.f, 0.f, 0.f};
#pragma unroll
  for (int mt = 0; mt < 2; ++mt)
#pragma unroll
    for (int j = 0; j < NT_PER_WAVE; ++j) acc[mt][j] = vzero;

  for (int d0 = 0; d0 < DIM; d0 += 32) {
    v16bf ahi[2], alo[2];
#pragma unroll
    for (int mt = 0; mt < 2; ++mt) {
      const int r = row0 + mt * 16 + lrow;
      ahi[mt] = load_frag_bf16(xh, r, d0, halfOff);
      alo[mt] = load_frag_bf16(xl, r, d0, halfOff);
    }
#pragma unroll
    for (int j = 0; j < NT_PER_WAVE; ++j) {
      const int n = (wave + j * WAVES) * 16 + lrow;
      const v16bf bhi = load_frag_bf16(ch, n, d0, halfOff);
      const v16bf blo = load_frag_bf16(cl, n, d0, halfOff);
#pragma unroll
      for (int mt = 0; mt < 2; ++mt) {
        acc[mt][j] = __builtin_amdgcn_wmma_f32_16x16x32_bf16(
            false, ahi[mt], false, bhi, (short)0, acc[mt][j], false, false);
        acc[mt][j] = __builtin_amdgcn_wmma_f32_16x16x32_bf16(
            false, ahi[mt], false, blo, (short)0, acc[mt][j], false, false);
        acc[mt][j] = __builtin_amdgcn_wmma_f32_16x16x32_bf16(
            false, alo[mt], false, bhi, (short)0, acc[mt][j], false, false);
      }
    }
  }

  // logits = cross - 0.5*||c_n||^2  (row-constant ||x||^2 cancels in log-softmax)
#pragma unroll
  for (int j = 0; j < NT_PER_WAVE; ++j) {
    const float cs = csq_g[(wave + j * WAVES) * 16 + lrow];
#pragma unroll
    for (int mt = 0; mt < 2; ++mt)
#pragma unroll
      for (int r = 0; r < 8; ++r)
        acc[mt][j][r] -= 0.5f * cs;
  }

  // row max over all 1024 centers
  float rmax[2][8];
#pragma unroll
  for (int mt = 0; mt < 2; ++mt)
#pragma unroll
    for (int r = 0; r < 8; ++r) {
      float m = acc[mt][0][r];
#pragma unroll
      for (int j = 1; j < NT_PER_WAVE; ++j) m = fmaxf(m, acc[mt][j][r]);
#pragma unroll
      for (int off = 1; off < 16; off <<= 1) m = fmaxf(m, __shfl_xor(m, off, 32));
      red[0][wave][mt * 16 + half * 8 + r] = m;
    }
  __syncthreads();
#pragma unroll
  for (int mt = 0; mt < 2; ++mt)
#pragma unroll
    for (int r = 0; r < 8; ++r) {
      const int m = mt * 16 + half * 8 + r;
      float v = red[0][0][m];
#pragma unroll
      for (int w = 1; w < WAVES; ++w) v = fmaxf(v, red[0][w][m]);
      rmax[mt][r] = v;
    }

  // sum of exp
  float rls[2][8];
#pragma unroll
  for (int mt = 0; mt < 2; ++mt)
#pragma unroll
    for (int r = 0; r < 8; ++r) {
      float s = 0.f;
#pragma unroll
      for (int j = 0; j < NT_PER_WAVE; ++j) s += __expf(acc[mt][j][r] - rmax[mt][r]);
#pragma unroll
      for (int off = 1; off < 16; off <<= 1) s += __shfl_xor(s, off, 32);
      red[1][wave][mt * 16 + half * 8 + r] = s;
    }
  __syncthreads();
#pragma unroll
  for (int mt = 0; mt < 2; ++mt)
#pragma unroll
    for (int r = 0; r < 8; ++r) {
      const int m = mt * 16 + half * 8 + r;
      float s = red[1][0][m];
#pragma unroll
      for (int w = 1; w < WAVES; ++w) s += red[1][w][m];
      rls[mt][r] = rmax[mt][r] + __logf(s);
    }

  // write log-softmax
#pragma unroll
  for (int mt = 0; mt < 2; ++mt)
#pragma unroll
    for (int j = 0; j < NT_PER_WAVE; ++j) {
      const int n0 = (wave + j * WAVES) * 16;
#pragma unroll
      for (int r = 0; r < 8; ++r) {
        const int m = mt * 16 + half * 8 + r;
        out[(size_t)(row0 + m) * DIM + n0 + lrow] = acc[mt][j][r] - rls[mt][r];
      }
    }
}

// ---------------------------------------------------------------------------
// Fallback (no workspace): fused on-the-fly conversion, csq with independent
// FMA partials (no serial add chain).
// ---------------------------------------------------------------------------
__device__ __forceinline__ void load16f(const float* __restrict__ p, float v[16]) {
  const float4 u0 = *(const float4*)(p + 0);
  const float4 u1 = *(const float4*)(p + 4);
  const float4 u2 = *(const float4*)(p + 16);
  const float4 u3 = *(const float4*)(p + 20);
  v[0]=u0.x;  v[1]=u0.y;  v[2]=u0.z;  v[3]=u0.w;
  v[4]=u1.x;  v[5]=u1.y;  v[6]=u1.z;  v[7]=u1.w;
  v[8]=u2.x;  v[9]=u2.y;  v[10]=u2.z; v[11]=u2.w;
  v[12]=u3.x; v[13]=u3.y; v[14]=u3.z; v[15]=u3.w;
}

__device__ __forceinline__ void split16(const float v[16], v16bf& hi, v16bf& lo) {
#pragma unroll
  for (int i = 0; i < 16; ++i) { __bf16 h, l; split1(v[i], h, l); hi[i] = h; lo[i] = l; }
}

__global__ __launch_bounds__(256)
void gemm_lsm_fused(const float* __restrict__ x,
                    const float* __restrict__ centers,
                    float* __restrict__ out) {
  __shared__ float red[2][WAVES][BLOCK_M];

  const int tid     = threadIdx.x;
  const int wave    = tid >> 5;
  const int lane    = tid & 31;
  const int lrow    = lane & 15;
  const int halfOff = (lane < 16) ? 0 : 8;
  const int half    = (lane < 16) ? 0 : 1;
  const int row0    = blockIdx.x * BLOCK_M;

  v8f acc[2][NT_PER_WAVE];
  const v8f vzero = {0.f, 0.f, 0.f, 0.f, 0.f, 0.f, 0.f, 0.f};
#pragma unroll
  for (int mt = 0; mt < 2; ++mt)
#pragma unroll
    for (int j = 0; j < NT_PER_WAVE; ++j) acc[mt][j] = vzero;

  float csqp[NT_PER_WAVE][4];
#pragma unroll
  for (int j = 0; j < NT_PER_WAVE; ++j)
#pragma unroll
    for (int i = 0; i < 4; ++i) csqp[j][i] = 0.f;

  for (int d0 = 0; d0 < DIM; d0 += 32) {
    v16bf ahi[2], alo[2];
#pragma unroll
    for (int mt = 0; mt < 2; ++mt) {
      float av[16];
      load16f(x + (size_t)(row0 + mt * 16 + lrow) * DIM + d0 + halfOff, av);
      split16(av, ahi[mt], alo[mt]);
    }
#pragma unroll
    for (int j = 0; j < NT_PER_WAVE; ++j) {
      const int n = (wave + j * WAVES) * 16 + lrow;
      float bv[16];
      load16f(centers + (size_t)n * DIM + d0 + halfOff, bv);
#pragma unroll
      for (int i = 0; i < 4; ++i)    // 4 independent FMA chains, not 1
        csqp[j][i] = fmaf(bv[i], bv[i],
                     fmaf(bv[4 + i], bv[4 + i],
                     fmaf(bv[8 + i], bv[8 + i],
                     fmaf(bv[12 + i], bv[12 + i], csqp[j][i]))));
      v16bf bhi, blo;
      split16(bv, bhi, blo);
#pragma unroll
      for (int mt = 0; mt < 2; ++mt) {
        acc[mt][j] = __builtin_amdgcn_wmma_f32_16x16x32_bf16(
            false, ahi[mt], false, bhi, (short)0, acc[mt][j], false, false);
        acc[mt][j] = __builtin_amdgcn_wmma_f32_16x16x32_bf16(
            false, ahi[mt], false, blo, (short)0, acc[mt][j], false, false);
        acc[mt][j] = __builtin_amdgcn_wmma_f32_16x16x32_bf16(
            false, alo[mt], false, bhi, (short)0, acc[mt][j], false, false);
      }
    }
  }

#pragma unroll
  for (int j = 0; j < NT_PER_WAVE; ++j) {
    float cs = (csqp[j][0] + csqp[j][1]) + (csqp[j][2] + csqp[j][3]);
    cs += __shfl_xor(cs, 16, 32);
#pragma unroll
    for (int mt = 0; mt < 2; ++mt)
#pragma unroll
      for (int r = 0; r < 8; ++r)
        acc[mt][j][r] -= 0.5f * cs;
  }

  float rmax[2][8];
#pragma unroll
  for (int mt = 0; mt < 2; ++mt)
#pragma unroll
    for (int r = 0; r < 8; ++r) {
      float m = acc[mt][0][r];
#pragma unroll
      for (int j = 1; j < NT_PER_WAVE; ++j) m = fmaxf(m, acc[mt][j][r]);
#pragma unroll
      for (int off = 1; off < 16; off <<= 1) m = fmaxf(m, __shfl_xor(m, off, 32));
      red[0][wave][mt * 16 + half * 8 + r] = m;
    }
  __syncthreads();
#pragma unroll
  for (int mt = 0; mt < 2; ++mt)
#pragma unroll
    for (int r = 0; r < 8; ++r) {
      const int m = mt * 16 + half * 8 + r;
      float v = red[0][0][m];
#pragma unroll
      for (int w = 1; w < WAVES; ++w) v = fmaxf(v, red[0][w][m]);
      rmax[mt][r] = v;
    }

  float rls[2][8];
#pragma unroll
  for (int mt = 0; mt < 2; ++mt)
#pragma unroll
    for (int r = 0; r < 8; ++r) {
      float s = 0.f;
#pragma unroll
      for (int j = 0; j < NT_PER_WAVE; ++j) s += __expf(acc[mt][j][r] - rmax[mt][r]);
#pragma unroll
      for (int off = 1; off < 16; off <<= 1) s += __shfl_xor(s, off, 32);
      red[1][wave][mt * 16 + half * 8 + r] = s;
    }
  __syncthreads();
#pragma unroll
  for (int mt = 0; mt < 2; ++mt)
#pragma unroll
    for (int r = 0; r < 8; ++r) {
      const int m = mt * 16 + half * 8 + r;
      float s = red[1][0][m];
#pragma unroll
      for (int w = 1; w < WAVES; ++w) s += red[1][w][m];
      rls[mt][r] = rmax[mt][r] + __logf(s);
    }

#pragma unroll
  for (int mt = 0; mt < 2; ++mt)
#pragma unroll
    for (int j = 0; j < NT_PER_WAVE; ++j) {
      const int n0 = (wave + j * WAVES) * 16;
#pragma unroll
      for (int r = 0; r < 8; ++r) {
        const int m = mt * 16 + half * 8 + r;
        out[(size_t)(row0 + m) * DIM + n0 + lrow] = acc[mt][j][r] - rls[mt][r];
      }
    }
}

// ---------------------------------------------------------------------------
extern "C" void kernel_launch(void* const* d_in, const int* in_sizes, int n_in,
                              void* d_out, int out_size, void* d_ws, size_t ws_size,
                              hipStream_t stream) {
  const float* x       = (const float*)d_in[0];
  const float* centers = (const float*)d_in[1];
  float* out           = (float*)d_out;
  const int batch = in_sizes[0] / DIM;                 // 16384
  const size_t xcnt = (size_t)batch * DIM;             // 16M
  const size_t ccnt = (size_t)DIM * DIM;               // 1M
  const size_t need = xcnt * 4 + ccnt * 4 + DIM * sizeof(float);  // hi+lo bf16 planes + csq

  if (d_ws != nullptr && ws_size >= need) {
    char* p = (char*)d_ws;
    __bf16* xh = (__bf16*)p;            p += xcnt * 2;
    __bf16* xl = (__bf16*)p;            p += xcnt * 2;
    __bf16* ch = (__bf16*)p;            p += ccnt * 2;
    __bf16* cl = (__bf16*)p;            p += ccnt * 2;
    float*  csq = (float*)p;

    split_x_kernel<<<dim3((unsigned)(xcnt / 8 / 256)), 256, 0, stream>>>(x, xh, xl);
    split_centers_kernel<<<dim3(DIM), 32, 0, stream>>>(centers, ch, cl, csq);
    gemm_lsm_bf16<<<dim3(batch / BLOCK_M), 256, 0, stream>>>(xh, xl, ch, cl, csq, out);
  } else {
    gemm_lsm_fused<<<dim3(batch / BLOCK_M), 256, 0, stream>>>(x, centers, out);
  }
}
